// Model_78091095376095
// MI455X (gfx1250) — matrix operand order repaired and checked by reading, not compiled, not run on hardware
//
#include <hip/hip_runtime.h>
#include <math.h>
#include <stdint.h>

#define TWO_PI_F 6.28318530717959f

typedef __bf16 bhalf;
typedef __attribute__((ext_vector_type(16))) __bf16 v16bf;
typedef __attribute__((ext_vector_type(8)))  float  v8f;

static __device__ inline float2 cmulf(float2 a, float2 b) {
  return make_float2(a.x * b.x - a.y * b.y, a.x * b.y + a.y * b.x);
}

// ---------------------------------------------------------------------------
// In-LDS Stockham radix-2 FFT. All threads of the block must call. Result
// ends in one of the two buffers; returned pointer says which. sign=-1 fwd.
// ---------------------------------------------------------------------------
static __device__ float2* lds_fft(float2* a, float2* b, int N, int sign, int tid, int nth) {
  int n = N, sstep = 1;
  float2* cur = a;
  float2* nxt = b;
  while (n > 1) {
    int m = n >> 1;
    for (int w = tid; w < (N >> 1); w += nth) {
      int p = w / sstep;
      int q = w - p * sstep;
      float2 x0 = cur[q + sstep * p];
      float2 x1 = cur[q + sstep * (p + m)];
      float2 sum = make_float2(x0.x + x1.x, x0.y + x1.y);
      float2 dif = make_float2(x0.x - x1.x, x0.y - x1.y);
      float ang = (float)sign * TWO_PI_F * (float)p / (float)n;
      float sn, cs;
      __sincosf(ang, &sn, &cs);
      nxt[q + sstep * 2 * p] = sum;
      nxt[q + sstep * (2 * p + 1)] = cmulf(dif, make_float2(cs, sn));
    }
    __syncthreads();
    float2* t = cur; cur = nxt; nxt = t;
    n = m; sstep <<= 1;
  }
  return cur;
}

// deterministic gaussian (stand-in for jax.random.normal(key 42))
static __device__ inline uint32_t pcg_hash(uint32_t v) {
  v = v * 747796405u + 2891336453u;
  uint32_t w = ((v >> ((v >> 28u) + 4u)) ^ v) * 277803737u;
  return (w >> 22u) ^ w;
}
static __device__ inline float gaussf(uint32_t idx) {
  uint32_t a = pcg_hash(idx * 2u + 1u);
  uint32_t b = pcg_hash(idx * 2u + 2u);
  float u1 = ((float)(a >> 8) + 0.5f) * (1.0f / 16777216.0f);
  float u2 = (float)(b >> 8) * (1.0f / 16777216.0f);
  return sqrtf(-2.0f * logf(u1)) * __cosf(TWO_PI_F * u2);
}

// ---------------------------------------------------------------------------
// Elementwise / small kernels
// ---------------------------------------------------------------------------
__global__ void k_zero(float* p, int n) {
  int i = blockIdx.x * 256 + threadIdx.x;
  if (i < n) p[i] = 0.f;
}
__global__ void k_copy(float* d, const float* s, int n) {
  int i = blockIdx.x * 256 + threadIdx.x;
  if (i < n) d[i] = s[i];
}
__global__ void k_add_colbias(float* x, const float* bias, int n, int total) {
  int i = blockIdx.x * 256 + threadIdx.x;
  if (i < total) x[i] += bias[i % n];
}

// STFT magnitude: one block per (frame, batch). mode 0: write, 1: subtract.
__global__ void k_stft_sub(const float* sig, float* spec, int mode) {
  int fr = blockIdx.x, b = blockIdx.y, tid = threadIdx.x;  // 256 threads
  __shared__ float2 A[2048];
  __shared__ float2 Bb[2048];
  for (int i = tid; i < 2048; i += 256) {
    int idx = fr * 256 + i;
    float v = (idx < 32768) ? sig[(size_t)b * 32768 + idx] : 0.f;
    float win = 0.5f - 0.5f * __cosf(TWO_PI_F * (float)i / 2047.f);
    A[i] = make_float2(v * win, 0.f);
  }
  __syncthreads();
  float2* R = lds_fft(A, Bb, 2048, -1, tid, 256);
  for (int bin = tid; bin < 1024; bin += 256) {
    float2 v = R[bin];
    float mag = sqrtf(v.x * v.x + v.y * v.y);
    float* p = spec + (size_t)b * 131072 + bin * 128 + fr;
    if (mode) *p -= mag; else *p = mag;
  }
}

// spec (b,c,t) -> x (c, b*128+t)
__global__ void k_spec_to_x(const float* spec, float* x) {
  int i = blockIdx.x * 256 + threadIdx.x;
  if (i >= 2 * 1024 * 128) return;
  int b = i / 131072, rem = i % 131072;
  int c = rem / 128, t = rem % 128;
  x[c * 256 + b * 128 + t] = spec[i];
}

// Xcat[2c][n] = x[c][n];  Xcat[2c+1][b,t] = x[c][b,t+d] (0 beyond T)
__global__ void k_build_xcat(const float* x, float* xc, int d) {
  int i = blockIdx.x * 256 + threadIdx.x;
  if (i >= 1024 * 256) return;
  int c = i >> 8, n = i & 255, t = n & 127;
  xc[(size_t)(2 * c) * 256 + n] = x[i];
  xc[(size_t)(2 * c + 1) * 256 + n] = (t + d < 128) ? x[i + d] : 0.f;
}

// ---------------------------------------------------------------------------
// bf16 WMMA GEMM: C[M,N] = A[M,K] @ B[K,N], f32 in/out, bf16 compute.
// REQUIRES: N % 128 == 0, K % 32 == 0 (true for all call sites).
// Block = 128 threads (4 wave32s): 16(M) x 128(N) tile, K stepped by 32.
// A is staged in LDS directly in fragment layout, contiguous k = 16h + s, so each lane
// reads its whole v16bf fragment as one contiguous 32B DS read (shared by
// all 4 waves). B is staged transposed (b128 global loads, packed bf16-pair
// DS stores) so per-lane fragments are contiguous. Each wave computes two
// 16x16 tiles (2 WMMAs, A-fragment reuse). Next K-slice of B is prefetched.
// ---------------------------------------------------------------------------
__global__ void __launch_bounds__(128) k_gemm_wmma(const float* __restrict__ A,
                                                   const float* __restrict__ B,
                                                   float* __restrict__ C,
                                                   int M, int N, int K) {
  __shared__ __attribute__((aligned(32))) bhalf sAF[32][16];   // [frag lane][elem]
  __shared__ __attribute__((aligned(32))) bhalf sBT[128][32];  // [col][k]
  int tid = threadIdx.x;
  int lane = tid & 31, wave = tid >> 5;
  int m0 = blockIdx.y * 16;
  int n0 = blockIdx.x * 128;
  int kh = lane >> 4;
  int ncol0 = n0 + wave * 32 + (lane & 15);
  v8f acc0 = {}, acc1 = {};
  // A staging geometry for this thread: fragment slot sl, 4 elems from e0
  int sl = tid >> 2;
  int e0 = (tid & 3) * 4;
  int srow = sl & 15, skh = sl >> 4;
  int ka0 = skh * 16 + e0;
  bool arow_ok = (m0 + srow) < M;
  for (int kk = 0; kk < K; kk += 32) {
    // ---- stage A fragment-layout (16x32 tile -> 32 lanes x 16 elems) ----
    float4 av = make_float4(0.f, 0.f, 0.f, 0.f);
    if (arow_ok) av = *(const float4*)(A + (size_t)(m0 + srow) * K + kk + ka0);
    sAF[sl][e0 + 0] = (bhalf)av.x;
    sAF[sl][e0 + 1] = (bhalf)av.y;
    sAF[sl][e0 + 2] = (bhalf)av.z;
    sAF[sl][e0 + 3] = (bhalf)av.w;
    // ---- stage B transposed (32 k-rows x 128 cols) ----
    // unit = (k-pair kp, 4-column group c4): 2x float4 loads, 4 packed stores
#pragma unroll
    for (int u = 0; u < 4; u++) {
      int q = u * 128 + tid;          // [0,512)
      int c4 = (q & 31) * 4;          // column group
      int kp = q >> 5;                // k-pair in [0,16)
      const float* bp = B + (size_t)(kk + 2 * kp) * N + n0 + c4;
      float4 r0 = *(const float4*)bp;
      float4 r1 = *(const float4*)(bp + N);
      sBT[c4 + 0][2 * kp] = (bhalf)r0.x; sBT[c4 + 0][2 * kp + 1] = (bhalf)r1.x;
      sBT[c4 + 1][2 * kp] = (bhalf)r0.y; sBT[c4 + 1][2 * kp + 1] = (bhalf)r1.y;
      sBT[c4 + 2][2 * kp] = (bhalf)r0.z; sBT[c4 + 2][2 * kp + 1] = (bhalf)r1.z;
      sBT[c4 + 3][2 * kp] = (bhalf)r0.w; sBT[c4 + 3][2 * kp + 1] = (bhalf)r1.w;
      if (kk + 32 < K) __builtin_prefetch(bp + (size_t)32 * N, 0, 1);
    }
    __syncthreads();
    v16bf af = *(const v16bf*)sAF[lane];
    v16bf bf0 = *(const v16bf*)&sBT[wave * 32 + (lane & 15)][kh * 16];
    v16bf bf1 = *(const v16bf*)&sBT[wave * 32 + 16 + (lane & 15)][kh * 16];
    acc0 = __builtin_amdgcn_wmma_f32_16x16x32_bf16(false, af, false, bf0,
                                                   (short)0, acc0, false, false);
    acc1 = __builtin_amdgcn_wmma_f32_16x16x32_bf16(false, af, false, bf1,
                                                   (short)0, acc1, false, false);
    __syncthreads();
  }
#pragma unroll
  for (int e = 0; e < 8; e++) {
    int m = m0 + e + (kh ? 8 : 0);
    if (m < M) {
      C[(size_t)m * N + ncol0] = acc0[e];
      C[(size_t)m * N + ncol0 + 16] = acc1[e];
    }
  }
}

// h = gelu(h + bias[row]); x += h; acc += h   (1024 x 256)
__global__ void k_post_enc(float* h, const float* bias, float* x, float* acc) {
  int i = blockIdx.x * 256 + threadIdx.x;
  if (i >= 1024 * 256) return;
  int o = i >> 8;
  float v = h[i] + bias[o];
  float g = 0.5f * v * (1.f + tanhf(0.7978845608f * (v + 0.044715f * v * v * v)));
  x[i] += g;
  acc[i] += g;
}

// out[M,N] = W[M,K] @ X[K,N] + bias[M]   (tiny M)
__global__ void k_linear_WX(float* out, const float* W, const float* X,
                            const float* bias, int M, int K, int N) {
  int i = blockIdx.x * 256 + threadIdx.x;
  if (i >= M * N) return;
  int m = i / N, n = i - m * N;
  float s = bias ? bias[m] : 0.f;
  for (int k = 0; k < K; k++) s += W[(size_t)m * K + k] * X[(size_t)k * N + n];
  out[i] = s;
}

// out[M,N] = X[M,K] @ W[K,N] + bias[N], activation 0:none 1:sigmoid 2:square
__global__ void k_linear(float* out, const float* X, const float* W,
                         const float* bias, int M, int K, int N, int act) {
  int i = blockIdx.x * 256 + threadIdx.x;
  if (i >= M * N) return;
  int m = i / N, n = i - m * N;
  float s = bias ? bias[n] : 0.f;
  for (int k = 0; k < K; k++) s += X[(size_t)m * K + k] * W[(size_t)k * N + n];
  if (act == 1) s = 1.f / (1.f + __expf(-s));
  else if (act == 2) s = s * s;
  out[i] = s;
}

// top-1 over relu(switch) per batch; emit vec + one-hot*val schedule
__global__ void k_pick(const float* evout, const float* swout,
                       float* vec_cur, float* sched_cur,
                       float* vecs_all, float* sched_all, int iter) {
  int b = blockIdx.x, t = threadIdx.x;  // 128 threads
  __shared__ float sval[128];
  __shared__ int sidx[128];
  float v = swout[b * 128 + t];
  v = v > 0.f ? v : 0.f;
  sval[t] = v; sidx[t] = t;
  __syncthreads();
  for (int off = 64; off >= 1; off >>= 1) {
    if (t < off) {
      float v2 = sval[t + off]; int i2 = sidx[t + off];
      if (v2 > sval[t] || (v2 == sval[t] && i2 < sidx[t])) { sval[t] = v2; sidx[t] = i2; }
    }
    __syncthreads();
  }
  int idx = sidx[0];
  float val = sval[0];
  if (t < 16) {
    float ev = evout[t * 256 + b * 128 + idx];
    vec_cur[b * 16 + t] = ev;
    vecs_all[(b * 16 + iter) * 16 + t] = ev;
  }
  float sv = (t == idx) ? val : 0.f;
  sched_cur[b * 128 + t] = sv;
  sched_all[(b * 16 + iter) * 128 + t] = sv;
}

// row reduction: mode 0 sum, mode 1 L2 norm
__global__ void k_rowred(const float* x, float* out, int n, int mode) {
  int r = blockIdx.x, t = threadIdx.x;  // 256 threads
  __shared__ float red[256];
  float s = 0.f;
  for (int i = t; i < n; i += 256) {
    float v = x[(size_t)r * n + i];
    s += mode ? v * v : v;
  }
  red[t] = s;
  __syncthreads();
  for (int off = 128; off >= 1; off >>= 1) {
    if (t < off) red[t] += red[t + off];
    __syncthreads();
  }
  if (t == 0) out[r] = mode ? sqrtf(red[0]) : red[0];
}

// x[r] = x[r]/(norm[r]+1e-8) * (amps ? amps[r] : 1)
__global__ void k_scale_norm(float* x, const float* norms, int n,
                             const float* amps, int total) {
  int i = blockIdx.x * 256 + threadIdx.x;
  if (i >= total) return;
  int r = i / n;
  float sc = 1.f / (norms[r] + 1e-8f);
  if (amps) sc *= amps[r];
  x[i] *= sc;
}

__global__ void k_softmax(float* x, int n) {
  int r = blockIdx.x, t = threadIdx.x;  // 256 threads
  __shared__ float red[256];
  float m = -1e30f;
  for (int i = t; i < n; i += 256) m = fmaxf(m, x[(size_t)r * n + i]);
  red[t] = m;
  __syncthreads();
  for (int off = 128; off >= 1; off >>= 1) {
    if (t < off) red[t] = fmaxf(red[t], red[t + off]);
    __syncthreads();
  }
  m = red[0];
  __syncthreads();
  float s = 0.f;
  for (int i = t; i < n; i += 256) {
    float e = __expf(x[(size_t)r * n + i] - m);
    x[(size_t)r * n + i] = e;
    s += e;
  }
  red[t] = s;
  __syncthreads();
  for (int off = 128; off >= 1; off >>= 1) {
    if (t < off) red[t] += red[t + off];
    __syncthreads();
  }
  float inv = 1.f / red[0];
  for (int i = t; i < n; i += 256) x[(size_t)r * n + i] *= inv;
}

// conv1d k=3 pad=1 over 128 channels, optional x2 nearest upsample of input,
// leaky_relu(0.2) epilogue
__global__ void k_conv1d_k3(float* out, const float* in, const float* w,
                            const float* bias, int BE, int Tout, int rep) {
  int i = blockIdx.x * 256 + threadIdx.x;
  int total = BE * 128 * Tout;
  if (i >= total) return;
  int t = i % Tout;
  int o = (i / Tout) % 128;
  int be = i / (Tout * 128);
  int Tin = rep ? (Tout >> 1) : Tout;
  float s = bias[o];
  const float* wb = w + (size_t)o * 128 * 3;
  const float* xb = in + (size_t)be * 128 * Tin;
  for (int c = 0; c < 128; c++) {
    for (int j = 0; j < 3; j++) {
      int ti = t + j - 1;
      if (ti < 0 || ti >= Tout) continue;
      int src = rep ? (ti >> 1) : ti;
      s += wb[c * 3 + j] * xb[(size_t)c * Tin + src];
    }
  }
  out[i] = s > 0.f ? s : 0.2f * s;
}

// filt[be,f,t] = sigmoid(W[f,:] . x[be,:,t] + b[f])^2 ; f<257, t<64
__global__ void k_filt(float* filt, const float* in, const float* W,
                       const float* bias, int BE) {
  int i = blockIdx.x * 256 + threadIdx.x;
  int total = BE * 257 * 64;
  if (i >= total) return;
  int t = i % 64;
  int f = (i / 64) % 257;
  int be = i / (64 * 257);
  float s = bias[f];
  for (int c = 0; c < 128; c++) s += W[(size_t)f * 128 + c] * in[(size_t)be * 128 * 64 + c * 64 + t];
  s = 1.f / (1.f + __expf(-s));
  filt[i] = s * s;
}

// per (frame,be): noise -> rfft -> *filt -> irfft
__global__ void k_noise_shape(float* shaped, const float* filt, int BE) {
  int fr = blockIdx.x, be = blockIdx.y, tid = threadIdx.x;  // 128 threads
  __shared__ float2 A[512];
  __shared__ float2 Bb[512];
  for (int i = tid; i < 512; i += 128) {
    float g = gaussf((uint32_t)((be * 64 + fr) * 512 + i));
    A[i] = make_float2(g, 0.f);
  }
  __syncthreads();
  float2* R = lds_fft(A, Bb, 512, -1, tid, 128);
  for (int k = tid; k < 512; k += 128) {
    int bin = (k <= 256) ? k : 512 - k;
    float f = filt[((size_t)be * 257 + bin) * 64 + fr];
    R[k].x *= f;
    R[k].y *= f;
  }
  __syncthreads();
  float2* other = (R == A) ? Bb : A;
  float2* R2 = lds_fft(R, other, 512, +1, tid, 128);
  for (int i = tid; i < 512; i += 128)
    shaped[((size_t)be * 64 + fr) * 512 + i] = R2[i].x * (1.f / 512.f);
}

// overlap-add (step 256, win 512) + envelope (64 -> 16384 linear interp)
__global__ void k_ola_env(float* imp, const float* shaped, const float* env64, int BE) {
  int i = blockIdx.x * 256 + threadIdx.x;
  if (i >= BE * 16384) return;
  int p = i % 16384, be = i / 16384;
  float s = 0.f;
  int f0 = p >> 8;
  for (int f = f0 - 1; f <= f0; f++) {
    if (f < 0 || f >= 64) continue;
    int j = p - f * 256;
    if (j >= 0 && j < 512) s += shaped[((size_t)be * 64 + f) * 512 + j];
  }
  float pos = (p + 0.5f) * (64.f / 16384.f) - 0.5f;
  int lo = (int)floorf(pos);
  if (lo < 0) lo = 0; if (lo > 63) lo = 63;
  int hi = lo + 1; if (hi > 63) hi = 63;
  float w = pos - (float)lo;
  w = fminf(fmaxf(w, 0.f), 1.f);
  float e = env64[be * 64 + lo] * (1.f - w) + env64[be * 64 + hi] * w;
  imp[i] = s * e;
}

// res *= interp(decay 128 -> 32768)
__global__ void k_decay_mul(float* res, const float* decay, int total) {
  int i = blockIdx.x * 256 + threadIdx.x;
  if (i >= total) return;
  int p = i % 32768, be = i / 32768;
  float pos = (p + 0.5f) * (128.f / 32768.f) - 0.5f;
  int lo = (int)floorf(pos);
  if (lo < 0) lo = 0; if (lo > 127) lo = 127;
  int hi = lo + 1; if (hi > 127) hi = 127;
  float w = pos - (float)lo;
  w = fminf(fmaxf(w, 0.f), 1.f);
  res[i] *= decay[be * 128 + lo] * (1.f - w) + decay[be * 128 + hi] * w;
}

__global__ void k_drywet(float* out, const float* imp, const float* wet,
                         const float* mix8, int total) {
  int i = blockIdx.x * 256 + threadIdx.x;
  if (i >= total) return;
  int p = i % 32768, be = i / 32768;
  const float* m = mix8 + be * 8;
  float dry = (p < 16384) ? imp[(size_t)be * 16384 + p] : 0.f;
  out[i] = dry * (m[0] + m[1] + m[2] + m[3]) + wet[i] * (m[4] + m[5] + m[6] + m[7]);
}

__global__ void k_build_up(float* up, const float* sched, int total) {
  int i = blockIdx.x * 256 + threadIdx.x;
  if (i >= total) return;
  int p = i % 32768, be = i / 32768;
  up[i] = ((p & 255) == 0) ? sched[be * 128 + (p >> 8)] : 0.f;
}

// c[b] = LN(mean_e unit_norm(vecs[b,e]))
__global__ void k_reverb_ctx(float* c, const float* vecs, int E,
                             const float* g, const float* bb) {
  int b = blockIdx.x;
  if (threadIdx.x != 0) return;
  float acc[16];
  for (int k = 0; k < 16; k++) acc[k] = 0.f;
  for (int e = 0; e < E; e++) {
    const float* v = vecs + (size_t)(b * E + e) * 16;
    float n = 0.f;
    for (int k = 0; k < 16; k++) n += v[k] * v[k];
    n = sqrtf(n) + 1e-8f;
    for (int k = 0; k < 16; k++) acc[k] += v[k] / n;
  }
  float mu = 0.f;
  for (int k = 0; k < 16; k++) { acc[k] /= (float)E; mu += acc[k]; }
  mu /= 16.f;
  float var = 0.f;
  for (int k = 0; k < 16; k++) { float d = acc[k] - mu; var += d * d; }
  float sd = sqrtf(var / 16.f + 1e-5f);
  for (int k = 0; k < 16; k++) c[b * 16 + k] = (acc[k] - mu) / sd * g[k] + bb[k];
}

__global__ void k_reverb_mix(float* out, const float* dry, const float* wet,
                             const float* mixv, int E, int total) {
  int i = blockIdx.x * 256 + threadIdx.x;
  if (i >= total) return;
  int b = (i / 32768) / E;
  out[i] = dry[i] * mixv[b * 2 + 0] + wet[i] * mixv[b * 2 + 1];
}

// ---------------------------------------------------------------------------
// 65536-pt FFT as four-step (256 x 256). Batched over grid.y.
// ---------------------------------------------------------------------------
__global__ void k_fft64k_p1(const float* rin, int L, const float2* cin,
                            float2* work, int sign) {
  int n1 = blockIdx.x, s = blockIdx.y, tid = threadIdx.x;  // 128 threads
  __shared__ float2 A[256];
  __shared__ float2 Bb[256];
  for (int j = tid; j < 256; j += 128) {
    float2 v;
    if (cin) {
      v = cin[(size_t)s * 65536 + n1 + 256 * j];
    } else {
      int idx = n1 + 256 * j;
      float r = (idx < L) ? rin[(size_t)s * L + idx] : 0.f;
      v = make_float2(r, 0.f);
    }
    A[j] = v;
  }
  __syncthreads();
  float2* R = lds_fft(A, Bb, 256, sign, tid, 128);
  float base = (float)sign * (TWO_PI_F / 65536.f) * (float)n1;
  for (int k2 = tid; k2 < 256; k2 += 128) {
    float sn, cs;
    __sincosf(base * (float)k2, &sn, &cs);
    work[(size_t)s * 65536 + n1 * 256 + k2] = cmulf(R[k2], make_float2(cs, sn));
  }
}

__global__ void k_fft64k_p2(const float2* work, float2* out, int sign, float scale) {
  int k2 = blockIdx.x, s = blockIdx.y, tid = threadIdx.x;  // 128 threads
  __shared__ float2 A[256];
  __shared__ float2 Bb[256];
  for (int n1 = tid; n1 < 256; n1 += 128)
    A[n1] = work[(size_t)s * 65536 + n1 * 256 + k2];
  __syncthreads();
  float2* R = lds_fft(A, Bb, 256, sign, tid, 128);
  for (int k1 = tid; k1 < 256; k1 += 128) {
    float2 v = R[k1];
    out[(size_t)s * 65536 + k2 + 256 * k1] = make_float2(v.x * scale, v.y * scale);
  }
}

__global__ void k_specmul(float2* A, const float2* B, int div, int total) {
  int i = blockIdx.x * 256 + threadIdx.x;
  if (i >= total) return;
  int s = i >> 16, k = i & 65535;
  A[i] = cmulf(A[i], B[(size_t)(s / div) * 65536 + k]);
}

__global__ void k_extract_real(const float2* in, float* out, int Lout, int total) {
  int i = blockIdx.x * 256 + threadIdx.x;
  if (i >= total) return;
  int s = i / Lout, n = i - s * Lout;
  out[i] = in[(size_t)s * 65536 + n].x;
}

// ---------------------------------------------------------------------------
// Host orchestration
// ---------------------------------------------------------------------------
struct Ctx {
  hipStream_t st;
  const float *audio, *enc_w, *enc_b, *ev_w, *ev_b, *sw_w, *sw_b;
  const float *fcw, *fcb, *iew, *ieb, *itsw, *itsb, *upsw, *upsb;
  const float *nw, *nb, *now, *nob, *waves, *atom_bias;
  const float *rsw, *rsb, *rmw, *rmb, *rdw, *rdb;
  const float *vlg, *vlb, *vrooms, *vroomw, *vroomb, *vmixw, *vmixb;
  float *spec, *encx, *encacc, *ench, *xcat, *evout, *swout;
  float *vecs_all, *sched_all, *vec_cur, *sched_cur, *ch;
  float *emb, *env64, *xa, *xb, *filt, *shaped, *imp_ws, *sel, *res, *decay;
  float *mix8, *amps, *norms, *wet, *mixed_ws, *up, *finalbuf;
  float *cctx, *roomsel, *room, *mixv;
  float2 *cA, *cT, *cB;
};

static inline int cdiv(int a, int b) { return (a + b - 1) / b; }

static void fftconv(Ctx& c, const float* a, int La, int batA,
                    const float* b, int Lb, int batB, int divb,
                    float* outReal, int Lout) {
  k_fft64k_p1<<<dim3(256, batA), 128, 0, c.st>>>(a, La, nullptr, c.cT, -1);
  k_fft64k_p2<<<dim3(256, batA), 128, 0, c.st>>>(c.cT, c.cA, -1, 1.f);
  k_fft64k_p1<<<dim3(256, batB), 128, 0, c.st>>>(b, Lb, nullptr, c.cT, -1);
  k_fft64k_p2<<<dim3(256, batB), 128, 0, c.st>>>(c.cT, c.cB, -1, 1.f);
  int tot = batA * 65536;
  k_specmul<<<cdiv(tot, 256), 256, 0, c.st>>>(c.cA, c.cB, divb, tot);
  k_fft64k_p1<<<dim3(256, batA), 128, 0, c.st>>>(nullptr, 0, c.cA, c.cT, +1);
  k_fft64k_p2<<<dim3(256, batA), 128, 0, c.st>>>(c.cT, c.cA, +1, 1.f / 65536.f);
  k_extract_real<<<cdiv(batA * Lout, 256), 256, 0, c.st>>>(c.cA, outReal, Lout, batA * Lout);
}

static void generate(Ctx& c, int E, const float* vecs, const float* sched,
                     float* out_final, float* out_imp, float* out_mixed,
                     float* out_amps) {
  int BE = 2 * E;
  // amps = sched.sum(-1)
  k_rowred<<<BE, 256, 0, c.st>>>(sched, out_amps, 128, 0);
  // emb
  k_linear<<<cdiv(BE * 256, 256), 256, 0, c.st>>>(c.emb, vecs, c.fcw, c.fcb, BE, 16, 256, 0);
  // impulse: env, start, upsample convs
  k_linear<<<cdiv(BE * 64, 256), 256, 0, c.st>>>(c.env64, c.emb, c.iew, c.ieb, BE, 256, 64, 2);
  k_linear<<<cdiv(BE * 512, 256), 256, 0, c.st>>>(c.xa, c.emb, c.itsw, c.itsb, BE, 256, 512, 0);
  float* pin = c.xa;
  float* pout = c.xb;
  int T = 4;
  for (int l = 0; l < 4; l++) {
    T *= 2;
    k_conv1d_k3<<<cdiv(BE * 128 * T, 256), 256, 0, c.st>>>(
        pout, pin, c.upsw + (size_t)l * 128 * 128 * 3, c.upsb + l * 128, BE, T, 1);
    float* t = pin; pin = pout; pout = t;
  }
  k_conv1d_k3<<<cdiv(BE * 128 * 64, 256), 256, 0, c.st>>>(pout, pin, c.nw, c.nb, BE, 64, 0);
  k_filt<<<cdiv(BE * 257 * 64, 256), 256, 0, c.st>>>(c.filt, pout, c.now, c.nob, BE);
  k_noise_shape<<<dim3(64, BE), 128, 0, c.st>>>(c.shaped, c.filt, BE);
  k_ola_env<<<cdiv(BE * 16384, 256), 256, 0, c.st>>>(out_imp, c.shaped, c.env64, BE);
  k_rowred<<<BE, 256, 0, c.st>>>(out_imp, c.norms, 16384, 1);
  k_scale_norm<<<cdiv(BE * 16384, 256), 256, 0, c.st>>>(out_imp, c.norms, 16384, nullptr, BE * 16384);
  // resonance selection + big WMMA GEMM against wavetable
  k_linear<<<cdiv(BE * 4096, 256), 256, 0, c.st>>>(c.sel, c.emb, c.rsw, c.rsb, BE, 256, 4096, 0);
  k_add_colbias<<<cdiv(BE * 4096, 256), 256, 0, c.st>>>(c.sel, c.atom_bias, 4096, BE * 4096);
  k_softmax<<<BE, 256, 0, c.st>>>(c.sel, 4096);
  k_gemm_wmma<<<dim3(cdiv(32768, 128), cdiv(BE, 16)), 128, 0, c.st>>>(c.sel, c.waves, c.res, BE, 32768, 4096);
  k_linear<<<cdiv(BE * 128, 256), 256, 0, c.st>>>(c.decay, c.emb, c.rdw, c.rdb, BE, 256, 128, 1);
  k_decay_mul<<<cdiv(BE * 32768, 256), 256, 0, c.st>>>(c.res, c.decay, BE * 32768);
  // wet = fft_convolve(imp_padded, res)
  fftconv(c, out_imp, 16384, BE, c.res, 32768, BE, 1, c.wet, 32768);
  k_linear<<<cdiv(BE * 8, 256), 256, 0, c.st>>>(c.mix8, c.emb, c.rmw, c.rmb, BE, 256, 8, 0);
  k_softmax<<<BE, 256, 0, c.st>>>(c.mix8, 8);
  k_drywet<<<cdiv(BE * 32768, 256), 256, 0, c.st>>>(out_mixed, out_imp, c.wet, c.mix8, BE * 32768);
  k_rowred<<<BE, 256, 0, c.st>>>(out_mixed, c.norms, 32768, 1);
  k_scale_norm<<<cdiv(BE * 32768, 256), 256, 0, c.st>>>(out_mixed, c.norms, 32768, out_amps, BE * 32768);
  // schedule placement + convolution
  k_build_up<<<cdiv(BE * 32768, 256), 256, 0, c.st>>>(c.up, sched, BE * 32768);
  fftconv(c, out_mixed, 32768, BE, c.up, 32768, BE, 1, c.finalbuf, 32768);
  // reverb
  k_reverb_ctx<<<2, 32, 0, c.st>>>(c.cctx, vecs, E, c.vlg, c.vlb);
  k_linear<<<1, 256, 0, c.st>>>(c.roomsel, c.cctx, c.vroomw, c.vroomb, 2, 16, 3, 0);
  k_softmax<<<2, 256, 0, c.st>>>(c.roomsel, 3);
  k_linear<<<cdiv(2 * 32768, 256), 256, 0, c.st>>>(c.room, c.roomsel, c.vrooms, nullptr, 2, 3, 32768, 0);
  k_linear<<<1, 256, 0, c.st>>>(c.mixv, c.cctx, c.vmixw, c.vmixb, 2, 16, 2, 0);
  k_softmax<<<2, 256, 0, c.st>>>(c.mixv, 2);
  fftconv(c, c.finalbuf, 32768, BE, c.room, 32768, 2, E, c.wet, 32768);
  k_reverb_mix<<<cdiv(BE * 32768, 256), 256, 0, c.st>>>(out_final, c.finalbuf, c.wet, c.mixv, E, BE * 32768);
}

extern "C" void kernel_launch(void* const* d_in, const int* in_sizes, int n_in,
                              void* d_out, int out_size, void* d_ws, size_t ws_size,
                              hipStream_t stream) {
  (void)in_sizes; (void)n_in; (void)out_size; (void)ws_size;
  Ctx c;
  c.st = stream;
  c.audio = (const float*)d_in[0];  c.enc_w = (const float*)d_in[1];
  c.enc_b = (const float*)d_in[2];  c.ev_w = (const float*)d_in[3];
  c.ev_b  = (const float*)d_in[4];  c.sw_w = (const float*)d_in[5];
  c.sw_b  = (const float*)d_in[6];  c.fcw  = (const float*)d_in[7];
  c.fcb   = (const float*)d_in[8];  c.iew  = (const float*)d_in[9];
  c.ieb   = (const float*)d_in[10]; c.itsw = (const float*)d_in[11];
  c.itsb  = (const float*)d_in[12]; c.upsw = (const float*)d_in[13];
  c.upsb  = (const float*)d_in[14]; c.nw   = (const float*)d_in[15];
  c.nb    = (const float*)d_in[16]; c.now  = (const float*)d_in[17];
  c.nob   = (const float*)d_in[18]; c.waves = (const float*)d_in[19];
  c.atom_bias = (const float*)d_in[20]; c.rsw = (const float*)d_in[21];
  c.rsb   = (const float*)d_in[22]; c.rmw  = (const float*)d_in[23];
  c.rmb   = (const float*)d_in[24]; c.rdw  = (const float*)d_in[25];
  c.rdb   = (const float*)d_in[26]; c.vlg  = (const float*)d_in[27];
  c.vlb   = (const float*)d_in[28]; c.vrooms = (const float*)d_in[29];
  c.vroomw = (const float*)d_in[30]; c.vroomb = (const float*)d_in[31];
  c.vmixw  = (const float*)d_in[32]; c.vmixb  = (const float*)d_in[33];

  float* w = (float*)d_ws;
  size_t off = 0;
  auto alloc = [&](size_t n) { float* p = w + off; off += n; return p; };
  c.spec = alloc(262144);    c.encx = alloc(262144);
  c.encacc = alloc(262144);  c.ench = alloc(262144);
  c.xcat = alloc(524288);    c.evout = alloc(4096);
  c.swout = alloc(256);      c.vecs_all = alloc(512);
  c.sched_all = alloc(4096); c.vec_cur = alloc(32);
  c.sched_cur = alloc(256);  c.ch = alloc(65536);
  c.emb = alloc(8192);       c.env64 = alloc(2048);
  c.xa = alloc(262144);      c.xb = alloc(262144);
  c.filt = alloc(526336);    c.shaped = alloc(1048576);
  c.imp_ws = alloc(524288);  c.sel = alloc(131072);
  c.res = alloc(1048576);    c.decay = alloc(4096);
  c.mix8 = alloc(256);       c.amps = alloc(32);
  c.norms = alloc(32);       c.wet = alloc(1048576);
  c.mixed_ws = alloc(1048576); c.up = alloc(1048576);
  c.finalbuf = alloc(1048576); c.cctx = alloc(32);
  c.roomsel = alloc(8);      c.room = alloc(65536);
  c.mixv = alloc(8);
  c.cA = (float2*)alloc(4194304);
  c.cT = (float2*)alloc(4194304);
  c.cB = (float2*)alloc(4194304);

  float* outF = (float*)d_out;
  float* o_final = outF;             // (2,16,32768)
  float* o_vecs  = outF + 1048576;   // (2,16,16)
  float* o_imp   = outF + 1049088;   // (2,16,16384)
  float* o_sched = outF + 1573376;   // (2,16,128)
  float* o_amps  = outF + 1577472;   // (2,16,1)
  float* o_mixed = outF + 1577504;   // (2,16,32768)

  // initial spectrogram
  k_stft_sub<<<dim3(128, 2), 256, 0, stream>>>(c.audio, c.spec, 0);

  const int dil[8] = {1, 2, 4, 8, 16, 32, 64, 1};
  for (int it = 0; it < 16; it++) {
    // --- encoder (8 dilated conv layers as WMMA GEMMs) ---
    k_spec_to_x<<<cdiv(262144, 256), 256, 0, stream>>>(c.spec, c.encx);
    k_zero<<<cdiv(262144, 256), 256, 0, stream>>>(c.encacc, 262144);
    for (int l = 0; l < 8; l++) {
      k_build_xcat<<<cdiv(262144, 256), 256, 0, stream>>>(c.encx, c.xcat, dil[l]);
      k_gemm_wmma<<<dim3(2, 64), 128, 0, stream>>>(
          c.enc_w + (size_t)l * 1024 * 2048, c.xcat, c.ench, 1024, 256, 2048);
      k_post_enc<<<cdiv(262144, 256), 256, 0, stream>>>(c.ench, c.enc_b + l * 1024, c.encx, c.encacc);
    }
    k_linear_WX<<<cdiv(16 * 256, 256), 256, 0, stream>>>(c.evout, c.ev_w, c.encacc, c.ev_b, 16, 1024, 256);
    k_linear_WX<<<1, 256, 0, stream>>>(c.swout, c.sw_w, c.encacc, c.sw_b, 1, 1024, 256);
    k_pick<<<2, 128, 0, stream>>>(c.evout, c.swout, c.vec_cur, c.sched_cur,
                                  c.vecs_all, c.sched_all, it);
    // --- generate one event per batch, render channel ---
    generate(c, 1, c.vec_cur, c.sched_cur, c.ch, c.imp_ws, c.mixed_ws, c.amps);
    // --- residual spectrogram ---
    k_stft_sub<<<dim3(128, 2), 256, 0, stream>>>(c.ch, c.spec, 1);
  }

  // final generate over all 16 events, writing straight into d_out regions
  generate(c, 16, c.vecs_all, c.sched_all, o_final, o_imp, o_mixed, o_amps);
  k_copy<<<cdiv(512, 256), 256, 0, stream>>>(o_vecs, c.vecs_all, 512);
  k_copy<<<cdiv(4096, 256), 256, 0, stream>>>(o_sched, c.sched_all, 4096);
}
